// DBA_5059471475307
// MI455X (gfx1250) — compile-verified, matrix operand order checked by reading, not run on hardware
//
#include <hip/hip_runtime.h>

typedef __attribute__((ext_vector_type(16))) _Float16 v16h;
typedef __attribute__((ext_vector_type(8)))  float    v8f;

#define SELU_SCALE 1.0507009873554805f
#define SELU_ALPHA 1.6732632423543772f
#define TAN30F     0.57735026918962576f
#define STRETCHF   0.78571428571428570f   /* 1/0.56 - 1 */
#define SPANF      1.1963f

__device__ __forceinline__ float selu_f(float x) {
  return SELU_SCALE * (x > 0.0f ? x : SELU_ALPHA * (expf(x) - 1.0f));
}

__device__ __forceinline__ void onera3(float x, float y, float z,
                                       float& ox, float& oy, float& oz) {
  float nx = x - TAN30F * y;
  float s  = 1.0f + STRETCHF * (y / SPANF);
  ox = nx * s; oy = y * s; oz = z * s;
}

// ---------------------------------------------------------------------------
// Pack W[K+?,Ncols] (f32, row-major) into WMMA B-fragment layout (f16):
//   Bpack[((kb * tiles_n + nt) * 32 + lane)] = v16h fragment for that wave slot.
// Element j of lane -> W row k = kb*32 + j + (j>=8 ? 8 : 0) + (lane>=16 ? 8 : 0),
// column nt*16 + (lane&15); out-of-range columns packed as 0 (branch-free).
// ---------------------------------------------------------------------------
__global__ void pack_b_kernel(const float* __restrict__ W, int ldw, int Ncols, int K,
                              _Float16* __restrict__ Bpack)
{
  const int tiles_n = (Ncols + 15) >> 4;
  const int total = (K >> 5) * tiles_n * 32;
  const int t = blockIdx.x * blockDim.x + threadIdx.x;
  if (t >= total) return;
  const int lane = t & 31;
  const int nt   = (t >> 5) % tiles_n;
  const int kbi  = (t >> 5) / tiles_n;
  const int hi   = lane >> 4;
  const int col  = nt * 16 + (lane & 15);
  const int colc = (col < Ncols) ? col : 0;
  const float mask = (col < Ncols) ? 1.0f : 0.0f;
  v16h b;
#pragma unroll
  for (int j = 0; j < 16; ++j) {
    int k = kbi * 32 + j + ((j >= 8) ? 8 : 0) + (hi ? 8 : 0);
    b[j] = (_Float16)(W[(long)k * ldw + colc] * mask);
  }
  ((v16h*)Bpack)[t] = b;
}

// ---------------------------------------------------------------------------
// WMMA GEMM: Out[M,Ncols] = X[M,K] @ W[0:K,:] (+ pos[M,3] @ W[K:K+3,:])
//            (+ bias) (selu optional).  One wave per 16x16 tile, f16 inputs /
//            f32 accumulate via v_wmma_f32_16x16x32_f16.  B comes pre-packed.
// A-frag (16-bit A 16x32, wave32): lane row = lane&15, hi lanes offset K by +8;
// element j -> k = j + (j>=8 ? 8 : 0) + (hi ? 8 : 0)  => two float4 pairs.
// C/D: acc[i] -> row mt*16 + i + 8*hi, col nt*16 + (lane&15).
// ---------------------------------------------------------------------------
__global__ void __launch_bounds__(256)
gemm_wmma_kernel(const float* __restrict__ X, int M, int K,
                 const _Float16* __restrict__ Bpack,
                 const float* __restrict__ W, int ldw, int Ncols,
                 const float* __restrict__ pos,
                 const float* __restrict__ bias, int do_selu,
                 float* __restrict__ Out)
{
  const int tiles_n = (Ncols + 15) >> 4;
  const int tiles_m = (M + 15) >> 4;
  const int wave = blockIdx.x * (blockDim.x >> 5) + (threadIdx.x >> 5);
  if (wave >= tiles_m * tiles_n) return;
  const int mt = wave / tiles_n;
  const int nt = wave - mt * tiles_n;
  const int lane = threadIdx.x & 31;
  const int hi   = lane >> 4;
  const int l15  = lane & 15;
  const int am   = mt * 16 + l15;
  const int amc  = am < M ? am : (M - 1);   // M is a multiple of 16 in this net
  const int bn   = nt * 16 + l15;

  const float* ap = X + (long)amc * K + (hi ? 8 : 0);
  const v16h* Bp  = (const v16h*)Bpack + (nt * 32 + lane);
  const int bstride = tiles_n * 32;

  v8f acc = {};
#pragma unroll 2
  for (int kb = 0; kb < K; kb += 32) {
    float4 x0 = ((const float4*)(ap + kb))[0];
    float4 x1 = ((const float4*)(ap + kb))[1];
    float4 x2 = ((const float4*)(ap + kb + 16))[0];
    float4 x3 = ((const float4*)(ap + kb + 16))[1];
    v16h b = Bp[(kb >> 5) * bstride];
    v16h a;
    a[0]=(_Float16)x0.x;  a[1]=(_Float16)x0.y;  a[2]=(_Float16)x0.z;  a[3]=(_Float16)x0.w;
    a[4]=(_Float16)x1.x;  a[5]=(_Float16)x1.y;  a[6]=(_Float16)x1.z;  a[7]=(_Float16)x1.w;
    a[8]=(_Float16)x2.x;  a[9]=(_Float16)x2.y;  a[10]=(_Float16)x2.z; a[11]=(_Float16)x2.w;
    a[12]=(_Float16)x3.x; a[13]=(_Float16)x3.y; a[14]=(_Float16)x3.z; a[15]=(_Float16)x3.w;
    acc = __builtin_amdgcn_wmma_f32_16x16x32_f16(
        /*neg_a=*/false, a, /*neg_b=*/false, b,
        /*c_mod=*/(short)0, acc, /*reuse_a=*/false, /*reuse_b=*/false);
  }

  float pw0 = 0.f, pw1 = 0.f, pw2 = 0.f, bs = 0.f;
  const int bnc = (bn < Ncols) ? bn : 0;
  if (pos) {
    pw0 = W[(long)(K + 0) * ldw + bnc];
    pw1 = W[(long)(K + 1) * ldw + bnc];
    pw2 = W[(long)(K + 2) * ldw + bnc];
  }
  if (bias) bs = bias[bnc];
#pragma unroll
  for (int i = 0; i < 8; ++i) {
    int m = mt * 16 + i + (hi ? 8 : 0);
    if (m < M && bn < Ncols) {
      float v = acc[i] + bs;
      if (pos) v += pos[3*m] * pw0 + pos[3*m + 1] * pw1 + pos[3*m + 2] * pw2;
      if (do_selu) v = selu_f(v);
      Out[(long)m * ldw + bn] = v;
    }
  }
}

// ---------------------------------------------------------------------------
// GCN degree / aggregation kernels
// ---------------------------------------------------------------------------
__global__ void deg_init_kernel(float* deg, int N) {
  int i = blockIdx.x * blockDim.x + threadIdx.x;
  if (i < N) deg[i] = 1.0f;                       // self loop
}
__global__ void deg_count_kernel(const int* __restrict__ dst, int E, float* deg) {
  int e = blockIdx.x * blockDim.x + threadIdx.x;
  if (e < E) atomicAdd(&deg[dst[e]], 1.0f);
}
__global__ void dinv_kernel(float* deg, int N) {
  int i = blockIdx.x * blockDim.x + threadIdx.x;
  if (i < N) deg[i] = rsqrtf(deg[i]);             // deg >= 1 always
}
__global__ void agg_init_kernel(const float* __restrict__ H, const float* __restrict__ dinv,
                                const float* __restrict__ bias, float* __restrict__ Out,
                                int N, int F) {
  long i = (long)blockIdx.x * blockDim.x + threadIdx.x;
  if (i >= (long)N * F) return;
  int v = (int)(i / F);
  int c = (int)(i - (long)v * F);
  float di = dinv[v];
  Out[i] = H[i] * di * di + bias[c];              // self-loop term + bias
}
__global__ void agg_edge_kernel(const float* __restrict__ H, const int* __restrict__ src,
                                const int* __restrict__ dst, int E,
                                const float* __restrict__ dinv, float* Out, int F) {
  long i = (long)blockIdx.x * blockDim.x + threadIdx.x;
  if (i >= (long)E * F) return;
  int e = (int)(i / F);
  int c = (int)(i - (long)e * F);
  int s = src[e], d = dst[e];
  atomicAdd(&Out[(long)d * F + c], H[(long)s * F + c] * dinv[s] * dinv[d]);
}
__global__ void selu_kernel(float* __restrict__ X, long n) {
  long i = (long)blockIdx.x * blockDim.x + threadIdx.x;
  if (i < n) X[i] = selu_f(X[i]);
}

// ---------------------------------------------------------------------------
// kNN (k=3) interpolate: search (LDS-chunked brute force) + gather
// ---------------------------------------------------------------------------
#define KNN_CHUNK 2048
__global__ void __launch_bounds__(256)
knn_search_kernel(const float* __restrict__ pos_src, int Ns,
                  const float* __restrict__ pos_tgt, int Nt,
                  int* __restrict__ idx_out, float* __restrict__ w_out)
{
  __shared__ float sx[KNN_CHUNK];
  __shared__ float sy[KNN_CHUNK];
  __shared__ float sz[KNN_CHUNK];
  const int t = blockIdx.x * blockDim.x + threadIdx.x;
  float tx = 0.f, ty = 0.f, tz = 0.f;
  if (t < Nt) onera3(pos_tgt[3*t], pos_tgt[3*t+1], pos_tgt[3*t+2], tx, ty, tz);
  float bd0 = 3.4e38f, bd1 = 3.4e38f, bd2 = 3.4e38f;
  int   bi0 = 0, bi1 = 0, bi2 = 0;
  for (int base = 0; base < Ns; base += KNN_CHUNK) {
    int cnt = min(KNN_CHUNK, Ns - base);
    __syncthreads();
    for (int i = threadIdx.x; i < cnt; i += blockDim.x) {
      float ox, oy, oz;
      onera3(pos_src[3*(base+i)], pos_src[3*(base+i)+1], pos_src[3*(base+i)+2], ox, oy, oz);
      sx[i] = ox; sy[i] = oy; sz[i] = oz;
    }
    __syncthreads();
    if (t < Nt) {
      for (int i = 0; i < cnt; ++i) {
        float dx = tx - sx[i], dy = ty - sy[i], dz = tz - sz[i];
        float d2 = dx*dx + dy*dy + dz*dz;
        if (d2 < bd2) {
          int gi = base + i;
          if (d2 < bd0)      { bd2=bd1; bi2=bi1; bd1=bd0; bi1=bi0; bd0=d2; bi0=gi; }
          else if (d2 < bd1) { bd2=bd1; bi2=bi1; bd1=d2;  bi1=gi; }
          else               { bd2=d2;  bi2=gi; }
        }
      }
    }
  }
  if (t < Nt) {
    float w0 = 1.0f / fmaxf(bd0, 1e-16f);
    float w1 = 1.0f / fmaxf(bd1, 1e-16f);
    float w2 = 1.0f / fmaxf(bd2, 1e-16f);
    float inv = 1.0f / (w0 + w1 + w2);
    idx_out[3*t] = bi0; idx_out[3*t+1] = bi1; idx_out[3*t+2] = bi2;
    w_out[3*t] = w0*inv; w_out[3*t+1] = w1*inv; w_out[3*t+2] = w2*inv;
  }
}

__global__ void knn_gather_kernel(const float* __restrict__ f, const int* __restrict__ idx,
                                  const float* __restrict__ w, float* __restrict__ out,
                                  int Nt, int F)
{
  long i = (long)blockIdx.x * blockDim.x + threadIdx.x;
  if (i >= (long)Nt * F) return;
  int t = (int)(i / F);
  int c = (int)(i - (long)t * F);
  out[i] = w[3*t]   * f[(long)idx[3*t]   * F + c]
         + w[3*t+1] * f[(long)idx[3*t+1] * F + c]
         + w[3*t+2] * f[(long)idx[3*t+2] * F + c];
}

// ---------------------------------------------------------------------------
extern "C" void kernel_launch(void* const* d_in, const int* in_sizes, int n_in,
                              void* d_out, int out_size, void* d_ws, size_t ws_size,
                              hipStream_t stream)
{
  (void)n_in; (void)out_size; (void)ws_size;
  const float* latent = (const float*)d_in[0];
  const float* pos0   = (const float*)d_in[1];
  const float* pos1   = (const float*)d_in[2];
  const float* pos2   = (const float*)d_in[3];
  const int*   e0 = (const int*)d_in[4]; const int E0 = in_sizes[4] / 2;
  const int*   e1 = (const int*)d_in[5]; const int E1 = in_sizes[5] / 2;
  const int*   e2 = (const int*)d_in[6]; const int E2 = in_sizes[6] / 2;
  const float* lin_w = (const float*)d_in[7];
  const float* lin_b = (const float*)d_in[8];
  const float* w0 = (const float*)d_in[9];  const float* b0 = (const float*)d_in[10];
  const float* w1 = (const float*)d_in[11]; const float* b1 = (const float*)d_in[12];
  const float* w2 = (const float*)d_in[13]; const float* b2 = (const float*)d_in[14];
  const float* w3 = (const float*)d_in[15]; const float* b3 = (const float*)d_in[16];
  const float* w4 = (const float*)d_in[17]; const float* b4 = (const float*)d_in[18];

  const int N0 = 2000, N1 = 8000, N2 = 32000, HID = 256, OUTF = 5;

  size_t off = 0;
  auto carve = [&](size_t bytes) -> void* {
    void* p = (void*)((char*)d_ws + off);
    off += (bytes + 255) & ~(size_t)255;
    return p;
  };
  float*     bufA  = (float*)carve((size_t)N2 * HID * 4);
  float*     bufB  = (float*)carve((size_t)N2 * HID * 4);
  float*     dinv0 = (float*)carve((size_t)N0 * 4);
  float*     dinv1 = (float*)carve((size_t)N1 * 4);
  float*     dinv2 = (float*)carve((size_t)N2 * 4);
  int*       kidx  = (int*)  carve((size_t)N2 * 3 * 4);
  float*     kw    = (float*)carve((size_t)N2 * 3 * 4);
  _Float16*  bpack = (_Float16*)carve((size_t)8 * 16 * 32 * 32);  // 128 KB max

  auto cdiv = [](long a, long b) { return (int)((a + b - 1) / b); };

  auto degrees = [&](const int* dst, int E, float* dinv, int N) {
    deg_init_kernel<<<cdiv(N, 256), 256, 0, stream>>>(dinv, N);
    deg_count_kernel<<<cdiv(E, 256), 256, 0, stream>>>(dst, E, dinv);
    dinv_kernel<<<cdiv(N, 256), 256, 0, stream>>>(dinv, N);
  };
  degrees(e0 + E0, E0, dinv0, N0);
  degrees(e1 + E1, E1, dinv1, N1);
  degrees(e2 + E2, E2, dinv2, N2);

  auto gemm = [&](const float* X, int M, int K, const float* W, int F,
                  const float* pos, const float* bias, int do_selu, float* Outp) {
    int tiles_n = cdiv(F, 16);
    int packN = (K >> 5) * tiles_n * 32;
    pack_b_kernel<<<cdiv(packN, 256), 256, 0, stream>>>(W, F, F, K, bpack);
    int tiles = cdiv(M, 16) * tiles_n;
    gemm_wmma_kernel<<<cdiv(tiles, 8), 256, 0, stream>>>(
        X, M, K, bpack, W, F, F, pos, bias, do_selu, Outp);
  };

  auto gcn = [&](const float* Xin, const float* pos, int N,
                 const int* esrc, const int* edst, int E,
                 const float* Wp, const float* Bp, const float* dinv,
                 float* Hbuf, float* Obuf, int F, int do_selu) {
    gemm(Xin, N, HID, Wp, F, pos, nullptr, 0, Hbuf);
    long nf = (long)N * F;
    agg_init_kernel<<<cdiv(nf, 256), 256, 0, stream>>>(Hbuf, dinv, Bp, Obuf, N, F);
    long ef = (long)E * F;
    agg_edge_kernel<<<cdiv(ef, 256), 256, 0, stream>>>(Hbuf, esrc, edst, E, dinv, Obuf, F);
    if (do_selu) selu_kernel<<<cdiv(nf, 256), 256, 0, stream>>>(Obuf, nf);
  };

  // encoder linear: x = selu(latent @ lin_w + lin_b)
  gemm(latent, N0, 64, lin_w, HID, nullptr, lin_b, 1, bufA);
  // level 0: two GCN layers
  gcn(bufA, pos0, N0, e0, e0 + E0, E0, w0, b0, dinv0, bufB, bufA, HID, 1);
  gcn(bufA, pos0, N0, e0, e0 + E0, E0, w1, b1, dinv0, bufB, bufA, HID, 1);
  // unpool 0 -> 1
  knn_search_kernel<<<cdiv(N1, 256), 256, 0, stream>>>(pos0, N0, pos1, N1, kidx, kw);
  knn_gather_kernel<<<cdiv((long)N1 * HID, 256), 256, 0, stream>>>(bufA, kidx, kw, bufB, N1, HID);
  // level 1
  gcn(bufB, pos1, N1, e1, e1 + E1, E1, w2, b2, dinv1, bufA, bufB, HID, 1);
  // unpool 1 -> 2
  knn_search_kernel<<<cdiv(N2, 256), 256, 0, stream>>>(pos1, N1, pos2, N2, kidx, kw);
  knn_gather_kernel<<<cdiv((long)N2 * HID, 256), 256, 0, stream>>>(bufB, kidx, kw, bufA, N2, HID);
  // level 2
  gcn(bufA, pos2, N2, e2, e2 + E2, E2, w3, b3, dinv2, bufB, bufA, HID, 1);
  // output layer (no selu)
  gcn(bufA, pos2, N2, e2, e2 + E2, E2, w4, b4, dinv2, bufB, (float*)d_out, OUTF, 0);
}
